// MaterialGraphModel_10496900071659
// MI455X (gfx1250) — matrix operand order repaired and checked by reading, not compiled, not run on hardware
//
#include <hip/hip_runtime.h>
#include <hip/hip_bf16.h>

typedef __bf16 bf16;
typedef __attribute__((ext_vector_type(16))) __bf16 v16bf;
typedef __attribute__((ext_vector_type(8)))  __bf16 v8bf;
typedef __attribute__((ext_vector_type(8)))  float  v8f;
typedef __attribute__((ext_vector_type(4)))  unsigned u32x4;
typedef __attribute__((ext_vector_type(8)))  unsigned u32x8;

#define MT 64
#define NT 128
#define KT 32

// ---------------------------------------------------------------------------
// A-operand source resolution.
// AMODE 0: plain bf16 matrix A[M,K]                       (staged via TDM)
// AMODE 1: edge msg input [h[src[e]] | h[tgt[e]] | e_bf[e]] (per-lane async DMA)
// AMODE 2: node upd input [h[n] | cvt(agg[n])]              (async + sync cast)
// K-chunks (32) never straddle the 256-element region boundaries.
// ---------------------------------------------------------------------------
template<int AMODE>
__device__ __forceinline__ const void* a_row_ptr(
    const bf16* A, const bf16* A2, const float* Agg,
    const int* srcI, const int* tgtI, int m, int K, int k)
{
  if (AMODE == 0) return (const void*)(A + (size_t)m * K + k);
  if (AMODE == 1) {
    if (k < 256) return (const void*)(A + (size_t)srcI[m] * 256 + k);
    if (k < 512) return (const void*)(A + (size_t)tgtI[m] * 256 + (k - 256));
    return (const void*)(A2 + (size_t)m * 256 + (k - 512));
  }
  if (k < 256) return (const void*)(A + (size_t)m * 256 + k);
  return (const void*)(Agg + (size_t)m * 256 + (k - 256));
}

// 16-byte asynchronous global -> LDS copy (ASYNCcnt path).
// LDS byte address = low 32 bits of the generic pointer (LDS aperture).
__device__ __forceinline__ void async_g2l_b128(void* lds, const void* gptr)
{
  unsigned loff = (unsigned)(size_t)lds;
  asm volatile("global_load_async_to_lds_b128 %0, %1, off"
               :: "v"(loff), "v"(gptr) : "memory");
}

// ---------------------------------------------------------------------------
// Weight pre-swizzle: f32 row-major W[K][Ncols] -> bf16 in this kernel's contiguous-k wave32 WMMA
// B-fragment order: [kChunk][colTile][lane][16 elems], where
//   lane = (col & 15) + 16 * ((k & 31) >> 4),  elem = k & 15.
// A lane's 16 B values become 32 contiguous bytes (two b128 global loads).
// ---------------------------------------------------------------------------
__global__ __launch_bounds__(256) void k_swz_b(
    const float* __restrict__ W, bf16* __restrict__ Wsz, int K, int Ncols)
{
  int idx = blockIdx.x * 256 + threadIdx.x;
  if (idx >= K * Ncols) return;
  int k = idx / Ncols, col = idx % Ncols;
  int c    = k >> 5;
  int kin  = k & 31;
  int t    = col >> 4;
  int lane = (col & 15) + ((kin >> 4) << 4);
  int elem = kin & 15;
  size_t o = (((size_t)c * (Ncols >> 4) + t) * 32 + lane) * 16 + elem;
  Wsz[o] = (bf16)W[idx];
}

// ---------------------------------------------------------------------------
// Tiled bf16 WMMA GEMM: C[M,Ncols] = act(A[M,K] @ B[K,Ncols] + bias)
// Block: 256 threads = 8 waves; tile 64(M) x 128(N); K in chunks of 32.
// A staged into double-buffered LDS (TDM for plain A, async DMA for gathers);
// B read directly from pre-swizzled fragment-ordered weights (L2-resident).
// EMODE 0: f32 store   1: bf16 store   2: both   3: atomicAdd into Csc[tgt[m]]
// ---------------------------------------------------------------------------
template<int AMODE, int EMODE, bool SILU>
__global__ __launch_bounds__(256) void k_gemm(
    const bf16* __restrict__ A, const bf16* __restrict__ A2, const float* __restrict__ Agg,
    const int* __restrict__ srcI, const int* __restrict__ tgtI,
    const bf16* __restrict__ Bsz, const float* __restrict__ bias,
    float* __restrict__ Cf, bf16* __restrict__ Cb, float* __restrict__ Csc,
    int M, int K, int Ncols)
{
  __shared__ alignas(16) bf16 lA[2][MT][KT + 8];   // double buffer; 80B row stride

  const int tid  = threadIdx.x;
  const int lane = tid & 31;
  const int wave = tid >> 5;
  const int mBase = blockIdx.x * MT;
  const int nBase = blockIdx.y * NT;

  // A-tile cooperative load coords (gather modes): 64 rows x 32 k, 8 bf16/thread
  const int tr = tid >> 2;
  const int tc = (tid & 3) * 8;
  int mL = mBase + tr; if (mL >= M) mL = M - 1;   // clamp partial last tile

  // B fragment pointer: advances by one kChunk stride per iteration
  const int nTiles = Ncols >> 4;
  const size_t bStride = (size_t)nTiles * 32 * 16;  // elems per kChunk
  const bf16* bp = Bsz + (((size_t)(nBase >> 4) + wave) * 32 + lane) * 16;

  // ---- stage one 64x32 A chunk into LDS buffer `buf` ----
  auto stage = [&](int k0, int buf) {
    if (AMODE == 0) {
      // Tensor Data Mover: one instruction DMAs the whole 64x32 tile, with
      // LDS padding (16 DWORDs data + 4 DWORDs pad = 80B row stride) and
      // tensor_dim OOB-zero handling of the partial last row tile.
      if (wave == 0) {
        unsigned lds_off = (unsigned)(size_t)(&lA[buf][0][0]);
        unsigned long long ga =
            (unsigned long long)(size_t)(A + (size_t)mBase * K + k0);
        u32x4 g0;
        g0[0] = 1u;                                           // count=1 (valid)
        g0[1] = lds_off;                                      // lds_addr
        g0[2] = (unsigned)ga;                                 // global_addr lo
        g0[3] = (unsigned)((ga >> 32) & 0x01FFFFFFu) | (2u << 30); // hi | type=2
        unsigned td0 = (unsigned)(K - k0);                    // cols remaining
        unsigned td1 = (unsigned)(M - mBase);                 // rows remaining
        u32x8 g1;
        g1[0] = (1u << 16) | (1u << 20) | (3u << 22) | (3u << 25);
                 // data_size=2B | pad_enable | pad_interval=16dw | pad_amount=4dw
        g1[1] = (td0 & 0xFFFFu) << 16;                        // tensor_dim0 lo
        g1[2] = (td0 >> 16) | ((td1 & 0xFFFFu) << 16);        // dim0 hi, dim1 lo
        g1[3] = (td1 >> 16) | (32u << 16);                    // dim1 hi, tile_dim0
        g1[4] = 64u;                                          // tile_dim1 (dim2=0)
        g1[5] = (unsigned)K;                                  // tensor_dim0_stride
        g1[6] = 0u;
        g1[7] = 0u;
        u32x4 gz = {0u, 0u, 0u, 0u};                          // groups 2/3 (2D)
        asm volatile("tensor_load_to_lds %0, %1, %2, %3"
                     :: "s"(g0), "s"(g1), "s"(gz), "s"(gz) : "memory");
      }
      return;
    }
    const void* p = a_row_ptr<AMODE>(A, A2, Agg, srcI, tgtI, mL, K, k0 + tc);
    if (AMODE == 2 && (k0 + tc) >= 256) {
      // fused f32 -> bf16 cast of agg: synchronous path (DMA cannot convert)
      const float4* pf = (const float4*)p;
      float4 f0 = pf[0], f1 = pf[1];
      v8bf t;
      t[0] = (bf16)f0.x; t[1] = (bf16)f0.y; t[2] = (bf16)f0.z; t[3] = (bf16)f0.w;
      t[4] = (bf16)f1.x; t[5] = (bf16)f1.y; t[6] = (bf16)f1.z; t[7] = (bf16)f1.w;
      *(v8bf*)&lA[buf][tr][tc] = t;
      if (k0 + KT < K)
        __builtin_prefetch(a_row_ptr<AMODE>(A, A2, Agg, srcI, tgtI, mL, K, k0 + KT + tc), 0, 1);
    } else {
      async_g2l_b128(&lA[buf][tr][tc], p);
    }
  };

  // ---- retire the in-flight transfer for the chunk about to be consumed ----
  auto waitStage = [&](bool last) {
    if (AMODE == 0) {
      if (wave == 0) {
        if (last) asm volatile("s_wait_tensorcnt 0x0" ::: "memory");
        else      asm volatile("s_wait_tensorcnt 0x1" ::: "memory");
      }
    } else if (AMODE == 1) {
      if (last) asm volatile("s_wait_asynccnt 0x0" ::: "memory");
      else      asm volatile("s_wait_asynccnt 0x1" ::: "memory");
    } else {
      asm volatile("s_wait_asynccnt 0x0" ::: "memory");  // mixed sync/async
    }
  };

  v8f acc[4] = {};

  // ---- one K-chunk of compute: B fragment + 4 M-subtile WMMAs ----
  auto compute = [&](int buf) {
    v8bf b0 = *(const v8bf*)bp;
    v8bf b1 = *(const v8bf*)(bp + 8);
    v16bf bfrag = __builtin_shufflevector(b0, b1,
        0, 1, 2, 3, 4, 5, 6, 7, 8, 9, 10, 11, 12, 13, 14, 15);
    bp += bStride;
    const int ko = (lane >= 16) ? 16 : 0;  // A layout: lanes>=16 hold K+16 halves
    #pragma unroll
    for (int s = 0; s < 4; ++s) {
      const int mrow = s * 16 + (lane & 15);
      v8bf alo = *(const v8bf*)&lA[buf][mrow][ko];        // K = ko .. ko+7
      v8bf ahi = *(const v8bf*)&lA[buf][mrow][8 + ko];    // K = 8+ko .. 15+ko
      v16bf afrag = __builtin_shufflevector(alo, ahi,
          0, 1, 2, 3, 4, 5, 6, 7, 8, 9, 10, 11, 12, 13, 14, 15);
      acc[s] = __builtin_amdgcn_wmma_f32_16x16x32_bf16(
          false, afrag, false, bfrag, (short)0, acc[s], false, false);
    }
  };

  // ---- double-buffered pipeline, last iteration peeled (branch-free waits) --
  const int nch = K / KT;
  stage(0, 0);
  for (int i = 0; i < nch - 1; ++i) {
    stage((i + 1) * KT, (i + 1) & 1);   // next chunk's DMA in flight
    waitStage(false);
    __syncthreads();                    // publish lA[i&1] to all waves
    compute(i & 1);
    __syncthreads();                    // reads done before buffer refill
  }
  waitStage(true);
  __syncthreads();
  compute((nch - 1) & 1);

  // ---- epilogue: bias (+SiLU), store / scatter ----
  const int col = nBase + wave * 16 + (lane & 15);
  const float bv = bias[col];
  const int rOff = (lane >= 16) ? 8 : 0;   // C/D layout: lanes>=16 hold M+8
  #pragma unroll
  for (int s = 0; s < 4; ++s) {
    #pragma unroll
    for (int r = 0; r < 8; ++r) {
      int m = mBase + s * 16 + r + rOff;
      if (m < M) {
        float v = acc[s][r] + bv;
        if (SILU) v = v / (1.f + __expf(-v));
        if (EMODE == 0) {
          Cf[(size_t)m * Ncols + col] = v;
        } else if (EMODE == 1) {
          Cb[(size_t)m * Ncols + col] = (bf16)v;
        } else if (EMODE == 2) {
          Cf[(size_t)m * Ncols + col] = v;
          Cb[(size_t)m * Ncols + col] = (bf16)v;
        } else {
          atomicAdd(&Csc[(size_t)tgtI[m] * Ncols + col], v);
        }
      }
    }
  }
}

// ---------------------------------------------------------------------------
// Residual + LayerNorm over H=256 (one row per 256-thread block, wave32 reduce)
// ---------------------------------------------------------------------------
__device__ __forceinline__ float waveSum(float v) {
  #pragma unroll
  for (int o = 16; o > 0; o >>= 1) v += __shfl_xor(v, o, 32);
  return v;
}

__global__ __launch_bounds__(256) void k_res_ln(
    const float* __restrict__ upd, const float* __restrict__ g, const float* __restrict__ b,
    float* __restrict__ h, bf16* __restrict__ hb, int N)
{
  __shared__ float red[8];
  const int n = blockIdx.x, t = threadIdx.x;
  float v = h[(size_t)n * 256 + t] + upd[(size_t)n * 256 + t];
  float s = waveSum(v);
  if ((t & 31) == 0) red[t >> 5] = s;
  __syncthreads();
  float mu = 0.f;
  #pragma unroll
  for (int i = 0; i < 8; ++i) mu += red[i];
  mu *= (1.f / 256.f);
  float d = v - mu;
  __syncthreads();
  float s2 = waveSum(d * d);
  if ((t & 31) == 0) red[t >> 5] = s2;
  __syncthreads();
  float var = 0.f;
  #pragma unroll
  for (int i = 0; i < 8; ++i) var += red[i];
  var *= (1.f / 256.f);
  float o = d * rsqrtf(var + 1e-5f) * g[t] + b[t];
  h[(size_t)n * 256 + t] = o;
  hb[(size_t)n * 256 + t] = (bf16)o;
}

// ---------------------------------------------------------------------------
// Segment mean/max pooling over sorted batch ids (run-length compressed atomics)
// ---------------------------------------------------------------------------
__device__ __forceinline__ unsigned fenc(float v) {
  unsigned u = __float_as_uint(v);
  return (u & 0x80000000u) ? ~u : (u | 0x80000000u);
}
__device__ __forceinline__ float fdec(unsigned u) {
  return __uint_as_float((u & 0x80000000u) ? (u & 0x7FFFFFFFu) : ~u);
}

__global__ __launch_bounds__(256) void k_pool(
    const float* __restrict__ h, const int* __restrict__ batch,
    float* __restrict__ sum, unsigned* __restrict__ mx, int* __restrict__ cnt, int N)
{
  const int t = threadIdx.x;
  int n0 = blockIdx.x * 64;
  int n1 = min(n0 + 64, N);
  int curg = -1, c = 0;
  float a = 0.f; unsigned m = 0u;
  for (int n = n0; n < n1; ++n) {
    int g = batch[n];
    if (g != curg) {
      if (curg >= 0) {
        atomicAdd(&sum[curg * 256 + t], a);
        atomicMax(&mx[curg * 256 + t], m);
        if (t == 0) atomicAdd(&cnt[curg], c);
      }
      curg = g; a = 0.f; m = 0u; c = 0;
    }
    float v = h[(size_t)n * 256 + t];
    a += v;
    unsigned e = fenc(v);
    m = (e > m) ? e : m;
    ++c;
  }
  if (curg >= 0) {
    atomicAdd(&sum[curg * 256 + t], a);
    atomicMax(&mx[curg * 256 + t], m);
    if (t == 0) atomicAdd(&cnt[curg], c);
  }
}

__global__ __launch_bounds__(256) void k_pool_fin(
    const float* __restrict__ sum, const unsigned* __restrict__ mx,
    const int* __restrict__ cnt,
    float* __restrict__ grout, bf16* __restrict__ grb)
{
  int g = blockIdx.x, t = threadIdx.x;
  int c = cnt[g];
  float mean = sum[g * 256 + t] / fmaxf((float)c, 1.f);
  float mv = (c > 0) ? fdec(mx[g * 256 + t]) : 0.f;
  grout[(size_t)g * 512 + t]       = mean;
  grout[(size_t)g * 512 + 256 + t] = mv;
  grb[(size_t)g * 512 + t]         = (bf16)mean;
  grb[(size_t)g * 512 + 256 + t]   = (bf16)mv;
}

__global__ __launch_bounds__(256) void k_cast(
    const float* __restrict__ in, bf16* __restrict__ out, int n)
{
  int i = blockIdx.x * 256 + threadIdx.x;
  if (i < n) out[i] = (bf16)in[i];
}

// ---------------------------------------------------------------------------
template<int AMODE, int EMODE, bool SILU>
static inline void gemm(const bf16* A, const bf16* A2, const float* Agg,
                        const int* srcI, const int* tgtI,
                        const bf16* Bsz, const float* bias,
                        float* Cf, bf16* Cb, float* Csc,
                        int M, int K, int Ncols, hipStream_t st)
{
  dim3 gr((unsigned)((M + MT - 1) / MT), (unsigned)(Ncols / NT));
  k_gemm<AMODE, EMODE, SILU><<<gr, 256, 0, st>>>(
      A, A2, Agg, srcI, tgtI, Bsz, bias, Cf, Cb, Csc, M, K, Ncols);
}

extern "C" void kernel_launch(void* const* d_in, const int* in_sizes, int n_in,
                              void* d_out, int out_size, void* d_ws, size_t ws_size,
                              hipStream_t stream)
{
  constexpr int N = 50000, E = 400000, NODE = 128, EDGE = 64, H = 256,
                LATENT = 128, L = 6, G = 64;

  const float* x     = (const float*)d_in[0];
  const int*   ei    = (const int*)d_in[1];
  const float* ea    = (const float*)d_in[2];
  const int*   bat   = (const int*)d_in[3];
  const float* np_W1 = (const float*)d_in[4];  const float* np_b1 = (const float*)d_in[5];
  const float* np_W2 = (const float*)d_in[6];  const float* np_b2 = (const float*)d_in[7];
  const float* ep_W1 = (const float*)d_in[8];  const float* ep_b1 = (const float*)d_in[9];
  const float* ep_W2 = (const float*)d_in[10]; const float* ep_b2 = (const float*)d_in[11];
  const float* mW1   = (const float*)d_in[12]; const float* mb1   = (const float*)d_in[13];
  const float* mW2   = (const float*)d_in[14]; const float* mb2   = (const float*)d_in[15];
  const float* uW1   = (const float*)d_in[16]; const float* ub1   = (const float*)d_in[17];
  const float* uW2   = (const float*)d_in[18]; const float* ub2   = (const float*)d_in[19];
  const float* lng   = (const float*)d_in[20]; const float* lnb   = (const float*)d_in[21];
  const float* op_W1 = (const float*)d_in[22]; const float* op_b1 = (const float*)d_in[23];
  const float* op_W2 = (const float*)d_in[24]; const float* op_b2 = (const float*)d_in[25];

  const int* srcI = ei;
  const int* tgtI = ei + E;

  char* wp = (char*)d_ws;
  auto alloc = [&](size_t bytes) -> void* {
    void* p = (void*)wp;
    wp += (bytes + 255) & ~(size_t)255;
    return p;
  };

  float*    h_f32 = (float*)alloc((size_t)N * H * 4);
  float*    upd_f = (float*)alloc((size_t)N * H * 4);
  float*    agg_f = (float*)alloc((size_t)N * H * 4);
  bf16*     h_bf  = (bf16*)alloc((size_t)N * H * 2);
  bf16*     e_bf  = (bf16*)alloc((size_t)E * H * 2);
  bf16*     te_bf = (bf16*)alloc((size_t)E * H * 2);
  bf16*     tn_bf = (bf16*)alloc((size_t)N * H * 2);
  bf16*     x_bf  = (bf16*)alloc((size_t)N * NODE * 2);
  bf16*     ea_bf = (bf16*)alloc((size_t)E * EDGE * 2);
  float*    psum  = (float*)alloc((size_t)G * H * 4);
  unsigned* pmax  = (unsigned*)alloc((size_t)G * H * 4);
  int*      pcnt  = (int*)alloc((size_t)G * 4);
  bf16*     gr_bf = (bf16*)alloc((size_t)G * 2 * H * 2);
  bf16*     tg_bf = (bf16*)alloc((size_t)G * H * 2);

  // Pre-swizzle all weights into WMMA B-fragment order (one-time, L2-resident)
  auto wswz = [&](const float* src, int K, int Ncols) -> bf16* {
    size_t n = (size_t)K * Ncols;
    bf16* dst = (bf16*)alloc(n * 2);
    k_swz_b<<<(unsigned)((n + 255) / 256), 256, 0, stream>>>(src, dst, K, Ncols);
    return dst;
  };
  bf16* npW1b = wswz(np_W1, NODE, H);
  bf16* npW2b = wswz(np_W2, H, H);
  bf16* epW1b = wswz(ep_W1, EDGE, H);
  bf16* epW2b = wswz(ep_W2, H, H);
  bf16* mW1b  = wswz(mW1, L * 3 * H, H);   // per-layer blocks stay contiguous
  bf16* mW2b  = wswz(mW2, L * H, H);
  bf16* uW1b  = wswz(uW1, L * 2 * H, H);
  bf16* uW2b  = wswz(uW2, L * H, H);
  bf16* opW1b = wswz(op_W1, 2 * H, H);
  bf16* opW2b = wswz(op_W2, H, LATENT);

  k_cast<<<(unsigned)(((size_t)N * NODE + 255) / 256), 256, 0, stream>>>(x, x_bf, N * NODE);
  k_cast<<<(unsigned)(((size_t)E * EDGE + 255) / 256), 256, 0, stream>>>(ea, ea_bf, E * EDGE);

  // node / edge projections
  gemm<0, 1, true >(x_bf,  nullptr, nullptr, nullptr, nullptr, npW1b, np_b1, nullptr, tn_bf, nullptr, N, NODE, H, stream);
  gemm<0, 2, false>(tn_bf, nullptr, nullptr, nullptr, nullptr, npW2b, np_b2, h_f32,  h_bf,  nullptr, N, H,    H, stream);
  gemm<0, 1, true >(ea_bf, nullptr, nullptr, nullptr, nullptr, epW1b, ep_b1, nullptr, te_bf, nullptr, E, EDGE, H, stream);
  gemm<0, 1, false>(te_bf, nullptr, nullptr, nullptr, nullptr, epW2b, ep_b2, nullptr, e_bf,  nullptr, E, H,    H, stream);

  // message-passing layers
  for (int l = 0; l < L; ++l) {
    hipMemsetAsync(agg_f, 0, (size_t)N * H * 4, stream);
    // msg = silu(cat(h[src],h[tgt],e) @ mW1 + mb1) @ mW2 + mb2; agg[tgt] += msg
    gemm<1, 1, true >(h_bf,  e_bf,   nullptr, srcI, tgtI,
                      mW1b + (size_t)l * 3 * H * H, mb1 + l * H,
                      nullptr, te_bf, nullptr, E, 3 * H, H, stream);
    gemm<0, 3, false>(te_bf, nullptr, nullptr, nullptr, tgtI,
                      mW2b + (size_t)l * H * H, mb2 + l * H,
                      nullptr, nullptr, agg_f, E, H, H, stream);
    // upd = silu(cat(h,agg) @ uW1 + ub1) @ uW2 + ub2; h = LN(h + upd)
    gemm<2, 1, true >(h_bf,  nullptr, agg_f, nullptr, nullptr,
                      uW1b + (size_t)l * 2 * H * H, ub1 + l * H,
                      nullptr, tn_bf, nullptr, N, 2 * H, H, stream);
    gemm<0, 0, false>(tn_bf, nullptr, nullptr, nullptr, nullptr,
                      uW2b + (size_t)l * H * H, ub2 + l * H,
                      upd_f, nullptr, nullptr, N, H, H, stream);
    k_res_ln<<<N, 256, 0, stream>>>(upd_f, lng + l * H, lnb + l * H, h_f32, h_bf, N);
  }

  // pooling + output MLP
  hipMemsetAsync(psum, 0, (size_t)G * H * 4, stream);
  hipMemsetAsync(pmax, 0, (size_t)G * H * 4, stream);
  hipMemsetAsync(pcnt, 0, (size_t)G * 4, stream);
  k_pool<<<(unsigned)((N + 63) / 64), 256, 0, stream>>>(h_f32, bat, psum, pmax, pcnt, N);

  float* out_f = (float*)d_out;
  k_pool_fin<<<G, 256, 0, stream>>>(psum, pmax, pcnt, out_f + (size_t)G * LATENT, gr_bf);

  gemm<0, 1, true >(gr_bf, nullptr, nullptr, nullptr, nullptr, opW1b, op_b1, nullptr, tg_bf, nullptr, G, 2 * H, H, stream);
  gemm<0, 0, false>(tg_bf, nullptr, nullptr, nullptr, nullptr, opW2b, op_b2, out_f,  nullptr, nullptr, G, H, LATENT, stream);
}
